// MSDeformableAttention_20950850470017
// MI455X (gfx1250) — matrix operand order repaired and checked by reading, not compiled, not run on hardware
//
#include <hip/hip_runtime.h>

// ---------------------------------------------------------------------------
// MS-Deformable-Attention for MI455X (gfx1250, wave32, WMMA).
//   bs=16, Lq=300, D=256, H=8, hd=32, levels {80x80,40x40,20x20,10x10}, Lv=8500
// All GEMMs use v_wmma_f32_16x16x32_f16 (fp32 accum, fp16 operands).
// ---------------------------------------------------------------------------

typedef __attribute__((ext_vector_type(16))) _Float16 v16h;
typedef __attribute__((ext_vector_type(8)))  _Float16 v8h;
typedef __attribute__((ext_vector_type(8)))  float    v8f;
typedef __attribute__((ext_vector_type(4)))  float    v4f;

#define K_DIM     256
#define WSH_PITCH 72   // 64 tile cols + 8 pad halves; row = 144 B (16B aligned)

__device__ inline v16h load16h(const _Float16* p) {
    v8h lo = *(const v8h*)(p);
    v8h hi = *(const v8h*)(p + 8);
    v16h r;
#pragma unroll
    for (int i = 0; i < 8; ++i) { r[i] = lo[i]; r[i + 8] = hi[i]; }
    return r;
}

// C[M][N] = A[M][256] * W[N][256]^T + bias[N]
// Block: 256 threads (8 waves). Block tile 128(M) x 64(N); wave tile 16 x 64.
__global__ __launch_bounds__(256) void gemm_f16wmma(
    const float* __restrict__ A,
    const float* __restrict__ W,
    const float* __restrict__ bias,
    float* __restrict__ C,
    int M, int N)
{
    __shared__ _Float16 Wsh[K_DIM][WSH_PITCH];   // Wsh[k][n] = W[colBase+n][k]

    const int colBase = blockIdx.y * 64;
    const int blockM  = blockIdx.x * 128;

    // Cooperative transpose of the 64x256 weight tile into LDS (fp32 -> f16).
    for (int idx = threadIdx.x; idx < 64 * K_DIM; idx += 256) {
        int n = idx >> 8;          // 0..63
        int k = idx & 255;         // consecutive threads -> consecutive k (coalesced)
        Wsh[k][n] = (_Float16)W[(size_t)(colBase + n) * K_DIM + k];
    }
    __syncthreads();

    const int wave = threadIdx.x >> 5;
    const int lane = threadIdx.x & 31;
    const int half = lane >> 4;     // K-split half for the A fragment
    const int mloc = lane & 15;     // A-fragment row within the 16x16 tile

    int rowA  = blockM + wave * 16 + mloc;
    int rowAc = rowA < M ? rowA : (M - 1);        // clamp: keep EXEC all-ones
    const float* arow = A + (size_t)rowAc * K_DIM;

    v8f acc0 = {}, acc1 = {}, acc2 = {}, acc3 = {};

#pragma unroll
    for (int k0 = 0; k0 < K_DIM; k0 += 32) {
        __builtin_prefetch(arow + k0 + 32, 0, 3);     // global_prefetch_b8

        // A fragment: 16-bit A 16x32 layout (ISA 7.12.2):
        //   lanes 0-15 : V0..V3 = K 0..7,  V4..V7 = K16..23
        //   lanes16-31 : V0..V3 = K 8..15, V4..V7 = K24..31
        int kA = k0 + half * 8;
        int kB = kA + 16;
        v4f f0 = *(const v4f*)(arow + kA);
        v4f f1 = *(const v4f*)(arow + kA + 4);
        v4f f2 = *(const v4f*)(arow + kB);
        v4f f3 = *(const v4f*)(arow + kB + 4);
        v16h a;
        a[0]  = (_Float16)f0[0]; a[1]  = (_Float16)f0[1];
        a[2]  = (_Float16)f0[2]; a[3]  = (_Float16)f0[3];
        a[4]  = (_Float16)f1[0]; a[5]  = (_Float16)f1[1];
        a[6]  = (_Float16)f1[2]; a[7]  = (_Float16)f1[3];
        a[8]  = (_Float16)f2[0]; a[9]  = (_Float16)f2[1];
        a[10] = (_Float16)f2[2]; a[11] = (_Float16)f2[3];
        a[12] = (_Float16)f3[0]; a[13] = (_Float16)f3[1];
        a[14] = (_Float16)f3[2]; a[15] = (_Float16)f3[3];

        // B fragments: 32x16 B layout, N = lane%16, K in same order as A (kA.., kB..).
        v16h b0, b1, b2, b3;
#pragma unroll
        for (int i = 0; i < 8; ++i) {
            b0[i] = Wsh[kA + i][ 0 + mloc]; b0[i + 8] = Wsh[kB + i][ 0 + mloc];
            b1[i] = Wsh[kA + i][16 + mloc]; b1[i + 8] = Wsh[kB + i][16 + mloc];
            b2[i] = Wsh[kA + i][32 + mloc]; b2[i + 8] = Wsh[kB + i][32 + mloc];
            b3[i] = Wsh[kA + i][48 + mloc]; b3[i + 8] = Wsh[kB + i][48 + mloc];
        }

        acc0 = __builtin_amdgcn_wmma_f32_16x16x32_f16(false, a, false, b0, (short)0, acc0, false, false);
        acc1 = __builtin_amdgcn_wmma_f32_16x16x32_f16(false, a, false, b1, (short)0, acc1, false, false);
        acc2 = __builtin_amdgcn_wmma_f32_16x16x32_f16(false, a, false, b2, (short)0, acc2, false, false);
        acc3 = __builtin_amdgcn_wmma_f32_16x16x32_f16(false, a, false, b3, (short)0, acc3, false, false);
    }

    // D layout: lanes 0-15 -> N=lane, M=r ; lanes 16-31 -> N=lane-16, M=8+r.
    const int n = mloc;
#pragma unroll
    for (int r = 0; r < 8; ++r) {
        int m    = half * 8 + r;
        int grow = blockM + wave * 16 + m;
        if (grow < M) {
            size_t base = (size_t)grow * N + colBase;
            C[base +  0 + n] = acc0[r] + bias[colBase +  0 + n];
            C[base + 16 + n] = acc1[r] + bias[colBase + 16 + n];
            C[base + 32 + n] = acc2[r] + bias[colBase + 32 + n];
            C[base + 48 + n] = acc3[r] + bias[colBase + 48 + n];
        }
    }
}

// ---------------------------------------------------------------------------
// Softmax + bilinear sampling + weighted sum.
// One wave per (b,q,h); lane = channel (hd=32 == wave32).
// ---------------------------------------------------------------------------
__device__ inline float samp_corner(const float* vb, int sv, int Wl, int Hl,
                                    int xi, int yi)
{
    bool valid = (xi >= 0) && (xi < Wl) && (yi >= 0) && (yi < Hl);
    int xc = min(max(xi, 0), Wl - 1);
    int yc = min(max(yi, 0), Hl - 1);
    float val = vb[(size_t)(sv + yc * Wl + xc) * 256];
    return valid ? val : 0.0f;
}

__global__ __launch_bounds__(256) void msda_sample(
    const float* __restrict__ rp,     // [bs*300][4]
    const float* __restrict__ off,    // [bs*300][256]  (H, 16 pts, xy)
    const float* __restrict__ lgts,   // [bs*300][128]  (H, 16 pts) raw logits
    const float* __restrict__ v,      // [bs*8500][256] projected value
    float* __restrict__ out)          // [bs*300][256]
{
    const int LVL_W[4]  = {80, 40, 20, 10};
    const int LVL_H[4]  = {80, 40, 20, 10};
    const int LVL_SV[4] = {0, 6400, 8000, 8400};

    int gw   = blockIdx.x * 8 + (threadIdx.x >> 5);  // global wave = (b,q,h)
    int lane = threadIdx.x & 31;                     // channel
    int h    = gw & 7;
    int bq   = gw >> 3;                              // b*300 + q
    int b    = bq / 300;

    const float* rpp = rp + (size_t)bq * 4;
    float cx = rpp[0], cy = rpp[1], sw = rpp[2], sh = rpp[3];

    const float* offp = off  + (size_t)bq * 256 + h * 32;
    const float* lg   = lgts + (size_t)bq * 128 + h * 16;

    // softmax over 16 logits (uniform across lanes; redundant compute is fine)
    float mx = lg[0];
#pragma unroll
    for (int p = 1; p < 16; ++p) mx = fmaxf(mx, lg[p]);
    float e[16];
    float s = 0.0f;
#pragma unroll
    for (int p = 0; p < 16; ++p) { e[p] = __expf(lg[p] - mx); s += e[p]; }
    float inv = 1.0f / s;

    const float* vb = v + (size_t)b * 8500 * 256 + h * 32 + lane;
    const float sc = 0.125f;   // (1/NUM_POINTS) * OFFSET_SCALE = 0.25 * 0.5

    float acc = 0.0f;
#pragma unroll
    for (int p = 0; p < 16; ++p) {
        int l  = p >> 2;
        int Wl = LVL_W[l], Hl = LVL_H[l], sv = LVL_SV[l];

        float lx = cx + offp[2 * p]     * sc * sw;
        float ly = cy + offp[2 * p + 1] * sc * sh;
        // grid = 2*loc-1 ; x = (grid+1)*W/2 - 0.5 = loc*W - 0.5
        float x = lx * (float)Wl - 0.5f;
        float y = ly * (float)Hl - 0.5f;
        float x0f = floorf(x), y0f = floorf(y);
        float wx = x - x0f, wy = y - y0f;
        int x0 = (int)x0f, y0 = (int)y0f;

        float v00 = samp_corner(vb, sv, Wl, Hl, x0,     y0);
        float v01 = samp_corner(vb, sv, Wl, Hl, x0 + 1, y0);
        float v10 = samp_corner(vb, sv, Wl, Hl, x0,     y0 + 1);
        float v11 = samp_corner(vb, sv, Wl, Hl, x0 + 1, y0 + 1);

        float bil = v00 * (1.0f - wx) * (1.0f - wy)
                  + v01 * wx          * (1.0f - wy)
                  + v10 * (1.0f - wx) * wy
                  + v11 * wx          * wy;
        acc += (e[p] * inv) * bil;
    }

    out[(size_t)bq * 256 + h * 32 + lane] = acc;
}

// ---------------------------------------------------------------------------
extern "C" void kernel_launch(void* const* d_in, const int* in_sizes, int n_in,
                              void* d_out, int out_size, void* d_ws, size_t ws_size,
                              hipStream_t stream)
{
    const float* query  = (const float*)d_in[0];   // [16,300,256]
    const float* refpts = (const float*)d_in[1];   // [16,300,1,4]
    const float* value  = (const float*)d_in[2];   // [16,8500,256]
    const float* W_off  = (const float*)d_in[3];   // [256,256]
    const float* b_off  = (const float*)d_in[4];   // [256]
    const float* W_attn = (const float*)d_in[5];   // [128,256]
    const float* b_attn = (const float*)d_in[6];   // [128]
    const float* W_val  = (const float*)d_in[7];   // [256,256]
    const float* b_val  = (const float*)d_in[8];   // [256]
    const float* W_out  = (const float*)d_in[9];   // [256,256]
    const float* b_out  = (const float*)d_in[10];  // [256]
    float* out = (float*)d_out;

    const int BS = 16, LQ = 300, LV = 8500;
    const int Mv = BS * LV;   // 136000 value rows
    const int Mq = BS * LQ;   // 4800 query rows

    char* ws = (char*)d_ws;
    float* v_ws    = (float*)ws;  ws += (size_t)Mv * 256 * sizeof(float);
    float* off_ws  = (float*)ws;  ws += (size_t)Mq * 256 * sizeof(float);
    float* attn_ws = (float*)ws;  ws += (size_t)Mq * 128 * sizeof(float);
    float* samp_ws = (float*)ws;

    dim3 blk(256);

    // 1) value projection: v = value @ W_val^T + b_val   (the big streaming GEMM)
    gemm_f16wmma<<<dim3((Mv + 127) / 128, 4), blk, 0, stream>>>(
        value, W_val, b_val, v_ws, Mv, 256);

    // 2) sampling offsets: off = query @ W_off^T + b_off
    gemm_f16wmma<<<dim3((Mq + 127) / 128, 4), blk, 0, stream>>>(
        query, W_off, b_off, off_ws, Mq, 256);

    // 3) attention logits: attn = query @ W_attn^T + b_attn (softmax fused later)
    gemm_f16wmma<<<dim3((Mq + 127) / 128, 2), blk, 0, stream>>>(
        query, W_attn, b_attn, attn_ws, Mq, 128);

    // 4) softmax + bilinear sample + weighted sum  (one wave per (b,q,h))
    msda_sample<<<dim3(Mq), blk, 0, stream>>>(
        refpts, off_ws, attn_ws, v_ws, samp_ws);

    // 5) output projection: out = sampled @ W_out^T + b_out
    gemm_f16wmma<<<dim3((Mq + 127) / 128, 4), blk, 0, stream>>>(
        samp_ws, W_out, b_out, out, Mq, 256);
}
